// TransformerEncoderLayer_5145370821113
// MI455X (gfx1250) — matrix operand order repaired and checked by reading, not compiled, not run on hardware
//
#include <hip/hip_runtime.h>

#define DM    1024
#define DFF   4096
#define NH    16
#define DH    64
#define SEQ   2048
#define BATCH 2
#define TOK   (BATCH*SEQ)   // 4096 tokens

typedef unsigned short u16;
typedef __bf16 v16bf __attribute__((ext_vector_type(16)));
typedef float  v8f   __attribute__((ext_vector_type(8)));
typedef int    v4i   __attribute__((vector_size(16)));

union BFrag { v16bf v; uint4 u[2]; u16 s[16]; };

#if defined(__HIP_DEVICE_COMPILE__) && __has_builtin(__builtin_amdgcn_global_load_async_to_lds_b128)
#define HAS_ASYNC_LDS 1
#else
#define HAS_ASYNC_LDS 0
#endif

__device__ __forceinline__ u16 f2bf(float f) {
  unsigned u = __float_as_uint(f);
  u += 0x7fffu + ((u >> 16) & 1u);   // round-to-nearest-even
  return (u16)(u >> 16);
}
__device__ __forceinline__ float bf2f(u16 h) {
  return __uint_as_float(((unsigned)h) << 16);
}
__device__ __forceinline__ v8f wmma_bf16(BFrag a, BFrag b, v8f c) {
  return __builtin_amdgcn_wmma_f32_16x16x32_bf16(
      false, a.v, false, b.v, (short)0, c, false, false);
}

// 16-byte global -> LDS copy: async DMA on CDNA5, sync fallback otherwise.
__device__ __forceinline__ void cp16(const u16* g, u16* l) {
#if HAS_ASYNC_LDS
  __builtin_amdgcn_global_load_async_to_lds_b128((v4i*)g, (v4i*)l, 0, 0);
#else
  *(uint4*)l = *(const uint4*)g;
#endif
}
__device__ __forceinline__ void async_wait_all() {
#if HAS_ASYNC_LDS
#if __has_builtin(__builtin_amdgcn_s_wait_asynccnt)
  __builtin_amdgcn_s_wait_asynccnt(0);
#else
  asm volatile("s_wait_asynccnt 0x0" ::: "memory");
#endif
#endif
}

// ---------------------------------------------------------------- packing ---
// W_qkv packed TRANSPOSED: out[n][k], n in [0,3072); Q part pre-scaled 1/8.
__global__ __launch_bounds__(256) void pack_qkv_kernel(
    const float* __restrict__ Wq, const float* __restrict__ Wk,
    const float* __restrict__ Wv, u16* __restrict__ out) {
  int idx = blockIdx.x * 256 + threadIdx.x;      // idx = n*1024 + k
  int n = idx >> 10, k = idx & 1023;
  const float* W = (n < 1024) ? Wq : ((n < 2048) ? Wk : Wv);
  float s = (n < 1024) ? 0.125f : 1.0f;          // fold 1/sqrt(dh) into Wq
  int nn = n & 1023, h = nn >> 6, e = nn & 63;
  out[idx] = f2bf(W[((size_t)(h * DM + k)) * DH + e] * s);
}

// f32 [K][N] -> bf16 transposed [N][K]
__global__ __launch_bounds__(256) void pack_matT_kernel(
    const float* __restrict__ in, u16* __restrict__ out, int K, int N) {
  int idx = blockIdx.x * 256 + threadIdx.x;      // idx = n*K + k
  int n = idx / K, k = idx % K;
  out[idx] = f2bf(in[(size_t)k * N + n]);
}

// -------------------------------------------------------------- layernorm ---
__global__ __launch_bounds__(256) void ln_kernel(
    const float* __restrict__ x, const float* __restrict__ g,
    const float* __restrict__ b, u16* __restrict__ out) {
  __shared__ float red[256];
  const int t = threadIdx.x, row = blockIdx.x;
  const float4 v = ((const float4*)(x + (size_t)row * DM))[t];
  red[t] = v.x + v.y + v.z + v.w;
  __syncthreads();
  for (int o = 128; o > 0; o >>= 1) { if (t < o) red[t] += red[t + o]; __syncthreads(); }
  const float mu = red[0] * (1.0f / DM);
  __syncthreads();
  const float dx = v.x - mu, dy = v.y - mu, dz = v.z - mu, dw = v.w - mu;
  red[t] = dx * dx + dy * dy + dz * dz + dw * dw;
  __syncthreads();
  for (int o = 128; o > 0; o >>= 1) { if (t < o) red[t] += red[t + o]; __syncthreads(); }
  const float rstd = rsqrtf(red[0] * (1.0f / DM) + 1e-5f);
  const float4 gv = ((const float4*)g)[t];
  const float4 bv = ((const float4*)b)[t];
  unsigned lo = (unsigned)f2bf(dx * rstd * gv.x + bv.x) |
                ((unsigned)f2bf(dy * rstd * gv.y + bv.y) << 16);
  unsigned hi = (unsigned)f2bf(dz * rstd * gv.z + bv.z) |
                ((unsigned)f2bf(dw * rstd * gv.w + bv.w) << 16);
  *(uint2*)&out[(size_t)row * DM + t * 4] = make_uint2(lo, hi);
}

// ------------------------------------------------------------- WMMA GEMM ----
// C[M,N] = A[M,K](bf16,row) @ Bt[N,K](bf16) [+bias] [GELU] [+res f32]
// Block 128x128, BK=64, 4 waves of 64x64, double-buffered async LDS staging.
// VSPLIT (QKV gemm): cols<2048 -> qk[row][col] (ld 2048); cols>=2048 ->
// vt[(col-2048)][row] (V stored transposed for attention staging).
template <bool OUT_BF16, bool GELU_ACT, bool RES, bool VSPLIT>
__global__ __launch_bounds__(128) void gemm_bf16_kernel(
    const u16* __restrict__ A, const u16* __restrict__ Bt,
    const float* __restrict__ bias, const float* __restrict__ res,
    void* __restrict__ outp, u16* __restrict__ vt, int M, int N, int K) {
  __shared__ __align__(16) u16 sA[2][128 * 72];   // [m][k]
  __shared__ __align__(16) u16 sB[2][128 * 72];   // [n][k]
  const int tid = threadIdx.x;
  const int m0 = blockIdx.y * 128, n0 = blockIdx.x * 128;
  const int lane = tid & 31, w = tid >> 5;
  const int wm = (w >> 1) * 64, wn = (w & 1) * 64;
  const int hi = lane >> 4, ln = lane & 15;

  v8f acc[4][4];
  v8f zero = {};
#pragma unroll
  for (int mi = 0; mi < 4; ++mi)
#pragma unroll
    for (int ni = 0; ni < 4; ++ni) acc[mi][ni] = zero;

  const int nk = K >> 6;
  // prologue stage
#pragma unroll
  for (int c = tid; c < 1024; c += 128) {
    int r = c >> 3, k8 = (c & 7) << 3;
    cp16(&A[(size_t)(m0 + r) * K + k8], &sA[0][r * 72 + k8]);
  }
#pragma unroll
  for (int c = tid; c < 1024; c += 128) {
    int r = c >> 3, k8 = (c & 7) << 3;
    cp16(&Bt[(size_t)(n0 + r) * K + k8], &sB[0][r * 72 + k8]);
  }
  async_wait_all();
  __syncthreads();

  for (int kt = 0; kt < nk; ++kt) {
    const int cur = kt & 1;
    if (kt + 1 < nk) {                       // prefetch next slab
      const int k0n = (kt + 1) << 6;
#pragma unroll
      for (int c = tid; c < 1024; c += 128) {
        int r = c >> 3, k8 = (c & 7) << 3;
        cp16(&A[(size_t)(m0 + r) * K + k0n + k8], &sA[cur ^ 1][r * 72 + k8]);
      }
#pragma unroll
      for (int c = tid; c < 1024; c += 128) {
        int r = c >> 3, k8 = (c & 7) << 3;
        cp16(&Bt[(size_t)(n0 + r) * K + k0n + k8], &sB[cur ^ 1][r * 72 + k8]);
      }
    }
#pragma unroll
    for (int ks = 0; ks < 2; ++ks) {
      BFrag af[4], bfr[4];
#pragma unroll
      for (int mi = 0; mi < 4; ++mi) {
        const u16* p = &sA[cur][(wm + mi * 16 + ln) * 72 + ks * 32 + hi * 8];
        af[mi].u[0] = *(const uint4*)p;
        af[mi].u[1] = *(const uint4*)(p + 16);
      }
#pragma unroll
      for (int ni = 0; ni < 4; ++ni) {
        const u16* p = &sB[cur][(wn + ni * 16 + ln) * 72 + ks * 32 + hi * 8];
        bfr[ni].u[0] = *(const uint4*)p;
        bfr[ni].u[1] = *(const uint4*)(p + 16);
      }
#pragma unroll
      for (int ni = 0; ni < 4; ++ni)
#pragma unroll
        for (int mi = 0; mi < 4; ++mi)
          acc[mi][ni] = wmma_bf16(af[mi], bfr[ni], acc[mi][ni]);
    }
    async_wait_all();
    __syncthreads();
  }

  const int ldo = VSPLIT ? 2048 : N;
#pragma unroll
  for (int mi = 0; mi < 4; ++mi) {
#pragma unroll
    for (int ni = 0; ni < 4; ++ni) {
      const int col = n0 + wn + ni * 16 + ln;
      const float bvv = bias ? bias[col] : 0.0f;
#pragma unroll
      for (int r = 0; r < 8; ++r) {
        const int row = m0 + wm + mi * 16 + hi * 8 + r;
        float v = acc[mi][ni][r] + bvv;
        if (GELU_ACT) v = 0.5f * v * (1.0f + erff(v * 0.70710678118654752f));
        if (RES) v += res[(size_t)row * N + col];
        if (VSPLIT) {
          if (col < 2048) ((u16*)outp)[(size_t)row * ldo + col] = f2bf(v);
          else vt[(size_t)(col - 2048) * TOK + row] = f2bf(v);
        } else if (OUT_BF16) {
          ((u16*)outp)[(size_t)row * N + col] = f2bf(v);
        } else {
          ((float*)outp)[(size_t)row * N + col] = v;
        }
      }
    }
  }
}

// --------------------------------------------------- flash attention (bf16) -
// grid: (SEQ/64, BATCH*NH); block: 128 threads = 4 waves, wave owns 16 q-rows.
// qk: [tok][2048] (Q|K, Q pre-scaled); vt: [1024][tok] (V transposed).
__global__ __launch_bounds__(128) void attention_kernel(
    const u16* __restrict__ qk, const u16* __restrict__ vt,
    u16* __restrict__ outO) {
  __shared__ __align__(16) u16 sQ[64 * 72];      // [qrow][e]
  __shared__ __align__(16) u16 sK[2][64 * 72];   // [key][e]
  __shared__ __align__(16) u16 sV[2][64 * 72];   // [e][key]
  __shared__ __align__(16) u16 sP[64 * 72];      // [qrow][key] per-wave rows
  const int tid = threadIdx.x;
  const int lane = tid & 31, w = tid >> 5;
  const int hi = lane >> 4, ln = lane & 15;
  const int q0 = blockIdx.x * 64;
  const int bh = blockIdx.y;
  const int hcol = (bh & (NH - 1)) * DH;
  const int tb = (bh >> 4) * SEQ;
  const int qr = w * 16;

#pragma unroll
  for (int c = tid; c < 512; c += 128) {         // stage Q once
    int r = c >> 3, e8 = (c & 7) << 3;
    cp16(&qk[(size_t)(tb + q0 + r) * 2048 + hcol + e8], &sQ[r * 72 + e8]);
  }
#pragma unroll
  for (int c = tid; c < 512; c += 128) {         // stage K tile 0
    int r = c >> 3, e8 = (c & 7) << 3;
    cp16(&qk[(size_t)(tb + r) * 2048 + 1024 + hcol + e8], &sK[0][r * 72 + e8]);
  }
#pragma unroll
  for (int c = tid; c < 512; c += 128) {         // stage V^T tile 0
    int r = c >> 3, t8 = (c & 7) << 3;
    cp16(&vt[(size_t)(hcol + r) * TOK + tb + t8], &sV[0][r * 72 + t8]);
  }
  async_wait_all();
  __syncthreads();

  float m[8], l[8];
  v8f O[4]; v8f zero = {};
#pragma unroll
  for (int r = 0; r < 8; ++r) { m[r] = -1e30f; l[r] = 0.0f; }
#pragma unroll
  for (int nt = 0; nt < 4; ++nt) O[nt] = zero;

  for (int j = 0; j < SEQ / 64; ++j) {
    const int cur = j & 1;
    if (j + 1 < SEQ / 64) {                      // prefetch next K/V tile
      const int kt = (j + 1) * 64;
#pragma unroll
      for (int c = tid; c < 512; c += 128) {
        int r = c >> 3, e8 = (c & 7) << 3;
        cp16(&qk[(size_t)(tb + kt + r) * 2048 + 1024 + hcol + e8],
             &sK[cur ^ 1][r * 72 + e8]);
      }
#pragma unroll
      for (int c = tid; c < 512; c += 128) {
        int r = c >> 3, t8 = (c & 7) << 3;
        cp16(&vt[(size_t)(hcol + r) * TOK + tb + kt + t8],
             &sV[cur ^ 1][r * 72 + t8]);
      }
    }

    // S(16x64) = Qrows @ K^T
    BFrag qa[2];
#pragma unroll
    for (int ks = 0; ks < 2; ++ks) {
      const u16* p = &sQ[(qr + ln) * 72 + ks * 32 + hi * 8];
      qa[ks].u[0] = *(const uint4*)p;
      qa[ks].u[1] = *(const uint4*)(p + 16);
    }
    v8f sacc[4];
#pragma unroll
    for (int nt = 0; nt < 4; ++nt) {
      sacc[nt] = zero;
#pragma unroll
      for (int ks = 0; ks < 2; ++ks) {
        BFrag kb;
        const u16* p = &sK[cur][(nt * 16 + ln) * 72 + ks * 32 + hi * 8];
        kb.u[0] = *(const uint4*)p;
        kb.u[1] = *(const uint4*)(p + 16);
        sacc[nt] = wmma_bf16(qa[ks], kb, sacc[nt]);
      }
    }

    // online softmax over this 64-key tile
    float lsum[8];
#pragma unroll
    for (int r = 0; r < 8; ++r) {
      float mv = sacc[0][r];
#pragma unroll
      for (int nt = 1; nt < 4; ++nt) mv = fmaxf(mv, sacc[nt][r]);
#pragma unroll
      for (int off = 8; off >= 1; off >>= 1) mv = fmaxf(mv, __shfl_xor(mv, off, 32));
      const float mnew = fmaxf(m[r], mv);
      const float alpha = __expf(m[r] - mnew);
      m[r] = mnew;
      l[r] *= alpha;
#pragma unroll
      for (int nt = 0; nt < 4; ++nt) O[nt][r] *= alpha;
      lsum[r] = 0.0f;
    }
#pragma unroll
    for (int nt = 0; nt < 4; ++nt) {
#pragma unroll
      for (int r = 0; r < 8; ++r) {
        const float p = __expf(sacc[nt][r] - m[r]);
        lsum[r] += p;
        sP[(qr + hi * 8 + r) * 72 + nt * 16 + ln] = f2bf(p);
      }
    }
#pragma unroll
    for (int r = 0; r < 8; ++r) {
      float s = lsum[r];
#pragma unroll
      for (int off = 8; off >= 1; off >>= 1) s += __shfl_xor(s, off, 32);
      l[r] += s;
    }
    asm volatile("s_wait_dscnt 0x0" ::: "memory");  // wave-local P round-trip

    // O(16x64) += P(16x64) @ V(64x64)
    BFrag pa[2];
#pragma unroll
    for (int ks = 0; ks < 2; ++ks) {
      const u16* p = &sP[(qr + ln) * 72 + ks * 32 + hi * 8];
      pa[ks].u[0] = *(const uint4*)p;
      pa[ks].u[1] = *(const uint4*)(p + 16);
    }
#pragma unroll
    for (int nt = 0; nt < 4; ++nt) {
#pragma unroll
      for (int ks = 0; ks < 2; ++ks) {
        BFrag vb;
        const u16* p = &sV[cur][(nt * 16 + ln) * 72 + ks * 32 + hi * 8];
        vb.u[0] = *(const uint4*)p;
        vb.u[1] = *(const uint4*)(p + 16);
        O[nt] = wmma_bf16(pa[ks], vb, O[nt]);
      }
    }
    async_wait_all();
    __syncthreads();
  }

#pragma unroll
  for (int nt = 0; nt < 4; ++nt) {               // head-concat layout
    const int col = hcol + nt * 16 + ln;
#pragma unroll
    for (int r = 0; r < 8; ++r) {
      const int row = tb + q0 + qr + hi * 8 + r;
      outO[(size_t)row * DM + col] = f2bf(O[nt][r] / l[r]);
    }
  }
}

// ------------------------------------------------------------------ launch --
extern "C" void kernel_launch(void* const* d_in, const int* in_sizes, int n_in,
                              void* d_out, int out_size, void* d_ws, size_t ws_size,
                              hipStream_t stream) {
  (void)in_sizes; (void)n_in; (void)out_size; (void)ws_size;
  const float* x   = (const float*)d_in[0];
  const float* Wq  = (const float*)d_in[1];
  const float* Wk  = (const float*)d_in[2];
  const float* Wv  = (const float*)d_in[3];
  const float* Wo  = (const float*)d_in[4];
  const float* bo  = (const float*)d_in[5];
  const float* g1  = (const float*)d_in[6];
  const float* be1 = (const float*)d_in[7];
  const float* g2  = (const float*)d_in[8];
  const float* be2 = (const float*)d_in[9];
  const float* W1  = (const float*)d_in[10];
  const float* b1  = (const float*)d_in[11];
  const float* W2  = (const float*)d_in[12];
  const float* b2  = (const float*)d_in[13];
  float* out = (float*)d_out;

  char* ws = (char*)d_ws;
  u16*   xn    = (u16*)(ws);                    //  8 MiB: LN1(x) bf16 [4096][1024]
  u16*   wqkvT = (u16*)(ws + (8ull  << 20));    //  6 MiB: [3072][1024] (Q rows ×1/8)
  u16*   woT   = (u16*)(ws + (14ull << 20));    //  2 MiB: Wo^T [1024][1024]
  u16*   w1T   = (u16*)(ws + (16ull << 20));    //  8 MiB: W1^T [4096][1024]
  u16*   w2T   = (u16*)(ws + (24ull << 20));    //  8 MiB: W2^T [1024][4096]
  u16*   qk    = (u16*)(ws + (32ull << 20));    // 16 MiB: [4096][2048] Q|K bf16
  u16*   vt    = (u16*)(ws + (48ull << 20));    //  8 MiB: V^T [1024][4096] bf16
  u16*   attnO = (u16*)(ws + (56ull << 20));    //  8 MiB: attn concat bf16
  float* hid   = (float*)(ws + (64ull << 20));  // 16 MiB: hidden f32
  u16*   hn    = (u16*)(ws + (80ull << 20));    //  8 MiB: LN2(hidden) bf16
  u16*   ff1   = (u16*)(ws + (88ull << 20));    // 32 MiB: GELU(hn@W1+b1) bf16

  pack_qkv_kernel<<<(DM * 3072) / 256, 256, 0, stream>>>(Wq, Wk, Wv, wqkvT);
  pack_matT_kernel<<<(DM * DM)  / 256, 256, 0, stream>>>(Wo, woT, DM, DM);
  pack_matT_kernel<<<(DM * DFF) / 256, 256, 0, stream>>>(W1, w1T, DM, DFF);
  pack_matT_kernel<<<(DFF * DM) / 256, 256, 0, stream>>>(W2, w2T, DFF, DM);

  // xn = LN1(x)
  ln_kernel<<<TOK, 256, 0, stream>>>(x, g1, be1, xn);
  // [Q|K] and V^T = xn @ [Wq/8 | Wk | Wv]
  gemm_bf16_kernel<true, false, false, true>
      <<<dim3(3072 / 128, TOK / 128), 128, 0, stream>>>(
          xn, wqkvT, nullptr, nullptr, qk, vt, TOK, 3072, DM);
  // attnO = softmax(QK^T)V, head-concat layout
  attention_kernel<<<dim3(SEQ / 64, BATCH * NH), 128, 0, stream>>>(qk, vt, attnO);
  // hidden = x + attnO @ Wo + bo
  gemm_bf16_kernel<false, false, true, false>
      <<<dim3(DM / 128, TOK / 128), 128, 0, stream>>>(
          attnO, woT, bo, x, hid, nullptr, TOK, DM, DM);
  // hn = LN2(hidden)
  ln_kernel<<<TOK, 256, 0, stream>>>(hid, g2, be2, hn);
  // ff1 = GELU(hn @ W1 + b1)
  gemm_bf16_kernel<true, true, false, false>
      <<<dim3(DFF / 128, TOK / 128), 128, 0, stream>>>(
          hn, w1T, b1, nullptr, ff1, nullptr, TOK, DFF, DM);
  // out = hidden + ff1 @ W2 + b2
  gemm_bf16_kernel<false, false, true, false>
      <<<dim3(DM / 128, TOK / 128), 128, 0, stream>>>(
          ff1, w2T, b2, hid, out, nullptr, TOK, DM, DFF);
}
